// SelfAttentionConv_62285615727127
// MI455X (gfx1250) — matrix operand order repaired and checked by reading, not compiled, not run on hardware
//
#include <hip/hip_runtime.h>
#include <hip/hip_bf16.h>

typedef __attribute__((ext_vector_type(16))) _Float16 v16h;
typedef __attribute__((ext_vector_type(8)))  _Float16 v8h;
typedef __attribute__((ext_vector_type(8)))  float    v8f;

#define BB 8
#define TT 1024
#define KK 128
#define HH 8
#define BH (BB*HH)
#define TOPK 64

// ---------------- workspace layout (bytes) ----------------
constexpr size_t OFF_X   = 0;                                  // x f16      (8*1024*128)
constexpr size_t OFF_WQT = OFF_X   + (size_t)2*BB*TT*KK;       // WqT f16    [5][1024][128]
constexpr size_t OFF_WKT = OFF_WQT + (size_t)2*5*1024*128;
constexpr size_t OFF_WVT = OFF_WKT + (size_t)2*5*1024*128;     // WvT f16    [1024][128]
constexpr size_t OFF_WUT = OFF_WVT + (size_t)2*1024*128;       // WuT f16    [128][1024]
constexpr size_t OFF_QP  = OFF_WUT + (size_t)2*128*1024;       // Qp f16     [64][1024][128]
constexpr size_t OFF_KP  = OFF_QP  + (size_t)2*BH*TT*KK;
constexpr size_t OFF_VT  = OFF_KP  + (size_t)2*BH*TT*KK;       // Vt f16     [64][128][1024]
constexpr size_t OFF_AT  = OFF_VT  + (size_t)2*BH*KK*TT;       // attn f16   [8192][1024]

// ---------------- helpers ----------------
__device__ inline v8f wmma16(v16h a, v16h b, v8f c) {
    return __builtin_amdgcn_wmma_f32_16x16x32_f16(false, a, false, b, (short)0, c, false, false);
}

__device__ inline v16h load_a_frag(const _Float16* base, int ld, int lane) {
    int m = lane & 15, half = lane >> 4;
    const _Float16* p = base + m * ld + half * 16;
    v8h lo = *(const v8h*)p;
    v8h hi = *(const v8h*)(p + 8);
    v16h r;
#pragma unroll
    for (int i = 0; i < 8; ++i) { r[i] = lo[i]; r[i + 8] = hi[i]; }
    return r;
}

__device__ inline int wave_sum_i(int v) {
#pragma unroll
    for (int o = 16; o > 0; o >>= 1) v += __shfl_xor(v, o, 32);
    return v;
}
__device__ inline float wave_sum_f(float v) {
#pragma unroll
    for (int o = 16; o > 0; o >>= 1) v += __shfl_xor(v, o, 32);
    return v;
}
__device__ inline float wave_max_f(float v) {
#pragma unroll
    for (int o = 16; o > 0; o >>= 1) v = fmaxf(v, __shfl_xor(v, o, 32));
    return v;
}
__device__ inline unsigned wave_min_u(unsigned v) {
#pragma unroll
    for (int o = 16; o > 0; o >>= 1) { unsigned t = __shfl_xor(v, o, 32); v = t < v ? t : v; }
    return v;
}

// monotonic float <-> sortable uint
__device__ inline unsigned f2ord(float f) {
    unsigned b = __float_as_uint(f);
    return (b & 0x80000000u) ? ~b : (b | 0x80000000u);
}
__device__ inline float ord2f(unsigned u) {
    return __uint_as_float((u & 0x80000000u) ? (u ^ 0x80000000u) : ~u);
}

// ---------------- prep kernels ----------------
__global__ void k_cvt(const float* __restrict__ src, _Float16* __restrict__ dst, int n) {
    int i = blockIdx.x * blockDim.x + threadIdx.x;
    if (i < n) dst[i] = (_Float16)src[i];
}

// Wq (1024,128,5) -> WqT[s][c][ci]
__global__ void k_wtrans(const float* __restrict__ w, _Float16* __restrict__ out, int n) {
    int i = blockIdx.x * blockDim.x + threadIdx.x;
    if (i < n) {
        int s = i / (1024 * 128);
        int rem = i % (1024 * 128);
        int c = rem >> 7, ci = rem & 127;
        out[i] = (_Float16)w[(c * 128 + ci) * 5 + s];
    }
}

// ---------------- QKV conv via shifted WMMA GEMMs ----------------
__global__ __launch_bounds__(256) void k_qkv(
    const _Float16* __restrict__ x16,
    const _Float16* __restrict__ WqT, const _Float16* __restrict__ WkT, const _Float16* __restrict__ WvT,
    const float* __restrict__ bq, const float* __restrict__ bk, const float* __restrict__ bv,
    _Float16* __restrict__ Qp, _Float16* __restrict__ Kp, _Float16* __restrict__ Vt)
{
    int wave = threadIdx.x >> 5, lane = threadIdx.x & 31;
    int tile = blockIdx.x * 8 + wave;          // 0..98303  (q | k | v, 32768 each)
    int kind = tile >> 15;
    int local = tile & 32767;
    int b  = local >> 12;                      // 4096 tiles per batch (64x64)
    int r2 = local & 4095;
    int t0 = (r2 >> 6) << 4;
    int c0 = (r2 & 63) << 4;
    int m = lane & 15, half = lane >> 4;
    const _Float16* xb = x16 + (size_t)b * TT * KK;
    v8f acc = {};

    if (kind < 2) {
        const _Float16* W = kind ? WkT : WqT;
#pragma unroll
        for (int s = 0; s < 5; ++s) {
            int trow = t0 + m + s - 4;         // causal left shift
#pragma unroll
            for (int c = 0; c < 4; ++c) {
                v16h a = {};
                if (trow >= 0) {
                    const _Float16* p = xb + trow * 128 + c * 32 + half * 16;
                    v8h lo = *(const v8h*)p; v8h hi = *(const v8h*)(p + 8);
#pragma unroll
                    for (int i = 0; i < 8; ++i) { a[i] = lo[i]; a[i + 8] = hi[i]; }
                }
                v16h bf = *(const v16h*)(W + (size_t)s * 1024 * 128 + (c0 + m) * 128 + c * 32 + half * 16);
                acc = wmma16(a, bf, acc);
            }
        }
        const float* bias = kind ? bk : bq;
        _Float16* outp = kind ? Kp : Qp;
        const float scale = 0.29730177875068026f;  // 128^-0.25
        int col = c0 + m;                           // D-layout: col = lane&15
        int ko = col >> 3, h = col & 7;
        float bc = bias[col];
#pragma unroll
        for (int r = 0; r < 8; ++r) {
            int t = t0 + r + 8 * half;
            outp[(((size_t)(b << 3) + h) * TT + t) * KK + ko] = (_Float16)((acc[r] + bc) * scale);
        }
    } else {
        // V: pointwise conv, stored transposed (bh, k, t) for the PV GEMM B-operand
#pragma unroll
        for (int c = 0; c < 4; ++c) {
            v16h a = load_a_frag(xb + t0 * 128 + c * 32, 128, lane);
            v16h bf = *(const v16h*)(WvT + (c0 + m) * 128 + c * 32 + half * 16);
            acc = wmma16(a, bf, acc);
        }
        int col = c0 + m;
        int ko = col >> 3, h = col & 7;
        float bc = bv[col];
#pragma unroll
        for (int r = 0; r < 8; ++r) {
            int t = t0 + r + 8 * half;
            Vt[(((size_t)(b << 3) + h) * KK + ko) * TT + t] = (_Float16)(acc[r] + bc);
        }
    }
}

// ---------------- fused attention: S=QK^T -> top-64 -> softmax -> O=PV ----------------
#define SST 1028   // f32 stride, padded (banks shift by 32 between row r and r+8)
#define PST 1032   // f16 stride, padded, keeps 16B alignment

__global__ __launch_bounds__(256) void k_attn(
    const _Float16* __restrict__ Qp, const _Float16* __restrict__ Kp,
    const _Float16* __restrict__ Vt, _Float16* __restrict__ attn)
{
    extern __shared__ char smem[];
    float*    S = (float*)smem;                        // 16 x SST f32
    _Float16* P = (_Float16*)(smem + 16 * SST * 4);    // 16 x PST f16

    int wave = threadIdx.x >> 5, lane = threadIdx.x & 31;
    int bh = blockIdx.x >> 6;
    int t0 = (blockIdx.x & 63) << 4;
    int m = lane & 15, half = lane >> 4;

    // ---- phase 1: S(16x1024) = Q_tile(16x128) @ K^T ----
    const _Float16* Qb = Qp + ((size_t)bh * TT + t0) * KK;
    const _Float16* Kb = Kp + (size_t)bh * TT * KK;
    v16h aQ[4];
#pragma unroll
    for (int c = 0; c < 4; ++c) aQ[c] = load_a_frag(Qb + c * 32, 128, lane);

    for (int nt = 0; nt < 8; ++nt) {
        int n0 = wave * 128 + nt * 16;
        v8f acc = {};
#pragma unroll
        for (int c = 0; c < 4; ++c) {
            v16h bf = *(const v16h*)(Kb + (n0 + m) * 128 + c * 32 + half * 16);
            acc = wmma16(aQ[c], bf, acc);
        }
#pragma unroll
        for (int r = 0; r < 8; ++r) S[(r + 8 * half) * SST + n0 + m] = acc[r];
    }
    __syncthreads();

    // ---- phase 2: per-row exact top-64 radix select + causal softmax ----
    for (int rr = 0; rr < 2; ++rr) {
        int row = wave * 2 + rr;
        const float* Sr = S + row * SST;
        unsigned u[32];
        unsigned umin = 0xFFFFFFFFu;
#pragma unroll
        for (int j = 0; j < 32; ++j) {
            unsigned uu = f2ord(Sr[lane + 32 * j]);
            u[j] = uu;
            umin = uu < umin ? uu : umin;
        }
        umin = wave_min_u(umin);

        // MSD radix-select of 64th-largest (over full row, pre-mask, per reference)
        unsigned prefix = 0;
        int want_k = TOPK;
        for (int bit = 31; bit >= 0; --bit) {
            unsigned bm = 1u << bit;
            unsigned mge = ~(bm - 1u);
            unsigned want = prefix | bm;
            int cnt = 0;
#pragma unroll
            for (int j = 0; j < 32; ++j) cnt += ((u[j] & mge) == want) ? 1 : 0;
            cnt = wave_sum_i(cnt);
            if (cnt >= want_k) prefix = want; else want_k -= cnt;
        }
        unsigned kth = prefix;
        float fmin = ord2f(umin);
        int tg = t0 + row;

        float mmax = -3.402823466e38f;
#pragma unroll
        for (int j = 0; j < 32; ++j) {
            int col = lane + 32 * j;
            if (col <= tg) {
                float f = (u[j] >= kth) ? ord2f(u[j]) : fmin;
                mmax = fmaxf(mmax, f);
            }
        }
        mmax = wave_max_f(mmax);

        float sum = 0.f;
#pragma unroll
        for (int j = 0; j < 32; ++j) {
            int col = lane + 32 * j;
            if (col <= tg) {
                float f = (u[j] >= kth) ? ord2f(u[j]) : fmin;
                sum += __expf(f - mmax);
            }
        }
        sum = wave_sum_f(sum);
        float inv = 1.0f / sum;

        _Float16* Pr = P + row * PST;
#pragma unroll
        for (int j = 0; j < 32; ++j) {
            int col = lane + 32 * j;
            float pv = 0.f;
            if (col <= tg) {
                float f = (u[j] >= kth) ? ord2f(u[j]) : fmin;
                pv = __expf(f - mmax) * inv;
            }
            Pr[col] = (_Float16)pv;
        }
    }
    __syncthreads();

    // ---- phase 3: O(16x128) = P(16x1024) @ V^T ; wave -> one 16-feature tile ----
    int n0 = wave * 16;
    const _Float16* Vb = Vt + (size_t)bh * KK * TT;
    v8f acc = {};
    for (int c = 0; c < 32; ++c) {
        v16h a = load_a_frag(P + c * 32, PST, lane);
        v16h bf = *(const v16h*)(Vb + (n0 + m) * TT + c * 32 + half * 16);
        acc = wmma16(a, bf, acc);
    }
    int b = bh >> 3, h = bh & 7;
#pragma unroll
    for (int r = 0; r < 8; ++r) {
        int t = t0 + r + 8 * half;
        attn[((size_t)b * TT + t) * (KK * HH) + h * KK + n0 + m] = (_Float16)acc[r];
    }
}

// ---------------- final projection: (8192x1024) @ Wu^T + bu ----------------
__global__ __launch_bounds__(256) void k_proj(
    const _Float16* __restrict__ attn, const _Float16* __restrict__ WuT,
    const float* __restrict__ bu, float* __restrict__ out)
{
    int wave = threadIdx.x >> 5, lane = threadIdx.x & 31;
    int tile = blockIdx.x * 8 + wave;   // 512 x 8 tiles
    int m0 = (tile >> 3) << 4;
    int n0 = (tile & 7) << 4;
    int m = lane & 15, half = lane >> 4;
    v8f acc = {};
    for (int c = 0; c < 32; ++c) {
        v16h a = load_a_frag(attn + (size_t)m0 * 1024 + c * 32, 1024, lane);
        v16h bf = *(const v16h*)(WuT + (n0 + m) * 1024 + c * 32 + half * 16);
        acc = wmma16(a, bf, acc);
    }
    float bb = bu[n0 + m];
#pragma unroll
    for (int r = 0; r < 8; ++r)
        out[(size_t)(m0 + r + 8 * half) * KK + n0 + m] = acc[r] + bb;
}

// ---------------- host launch ----------------
extern "C" void kernel_launch(void* const* d_in, const int* in_sizes, int n_in,
                              void* d_out, int out_size, void* d_ws, size_t ws_size,
                              hipStream_t stream) {
    const float* x  = (const float*)d_in[0];
    const float* Wq = (const float*)d_in[1];
    const float* bq = (const float*)d_in[2];
    const float* Wk = (const float*)d_in[3];
    const float* bk = (const float*)d_in[4];
    const float* Wv = (const float*)d_in[5];
    const float* bv = (const float*)d_in[6];
    const float* Wu = (const float*)d_in[7];
    const float* bu = (const float*)d_in[8];

    char* ws = (char*)d_ws;
    _Float16* x16 = (_Float16*)(ws + OFF_X);
    _Float16* WqT = (_Float16*)(ws + OFF_WQT);
    _Float16* WkT = (_Float16*)(ws + OFF_WKT);
    _Float16* WvT = (_Float16*)(ws + OFF_WVT);
    _Float16* WuT = (_Float16*)(ws + OFF_WUT);
    _Float16* Qp  = (_Float16*)(ws + OFF_QP);
    _Float16* Kp  = (_Float16*)(ws + OFF_KP);
    _Float16* Vt  = (_Float16*)(ws + OFF_VT);
    _Float16* At  = (_Float16*)(ws + OFF_AT);

    const int NX = BB * TT * KK;        // 1048576
    const int NW = 1024 * 128;          // 131072
    const int NQW = 5 * 1024 * 128;     // 655360

    k_cvt<<<(NX + 255) / 256, 256, 0, stream>>>(x, x16, NX);
    k_cvt<<<(NW + 255) / 256, 256, 0, stream>>>(Wv, WvT, NW);
    k_cvt<<<(NW + 255) / 256, 256, 0, stream>>>(Wu, WuT, NW);
    k_wtrans<<<(NQW + 255) / 256, 256, 0, stream>>>(Wq, WqT, NQW);
    k_wtrans<<<(NQW + 255) / 256, 256, 0, stream>>>(Wk, WkT, NQW);

    k_qkv<<<12288, 256, 0, stream>>>(x16, WqT, WkT, WvT, bq, bk, bv, Qp, Kp, Vt);

    size_t lds = (size_t)16 * SST * 4 + (size_t)16 * PST * 2;   // 98816 B
    k_attn<<<BH * (TT / 16), 256, lds, stream>>>(Qp, Kp, Vt, At);

    k_proj<<<512, 256, 0, stream>>>(At, WuT, bu, (float*)d_out);
}
